// GeneralNonLinearBiasReadoutBlock_86835648790774
// MI455X (gfx1250) — matrix operand order repaired and checked by reading, not compiled, not run on hardware
//
#include <hip/hip_runtime.h>

// ---------------- types ----------------
typedef __bf16 bf16_t;
typedef __attribute__((ext_vector_type(16))) __bf16   bf16x16;
typedef __attribute__((ext_vector_type(8)))  float    v8f;
typedef __attribute__((ext_vector_type(4)))  unsigned int v4u;

union FragU { bf16x16 v; v4u q[2]; };

// ---------------- constants ----------------
#define N_ROWS     65536
#define MUL_IN     512
#define MUL_H      256
#define A_IN       0.0441941738241592f   /* 512^-0.5 */
#define A_H        0.0625f               /* 256^-0.5 */
#define C_SILU     1.6765688f
#define C_SIG      1.8462723f

#define ROWS_PER_WG 32
#define X_PITCH    520                    /* 512 + 8 pad (bf16 elems) */
#define X_PLANE    (ROWS_PER_WG * X_PITCH)
#define H_PITCH    264                    /* 256 + 8 pad */
#define H_PLANE    (ROWS_PER_WG * H_PITCH)

__device__ __forceinline__ float sigmoid_f(float z) { return 1.0f / (1.0f + __expf(-z)); }
__device__ __forceinline__ float silu_f(float z)    { return z * sigmoid_f(z); }

// ---- A fragment: 16x32 bf16, row-major in LDS (row = row16 + lane&15) ----
// lane<16 : K = k0..k0+15;  lane>=16 : K = k0+16..k0+31
__device__ __forceinline__ bf16x16 load_a(const bf16_t* __restrict__ base, int pitch,
                                          int row16, int k0, int lane) {
  int row = row16 + (lane & 15);
  int kb  = k0 + ((lane >> 4) << 4);
  FragU f;
  f.q[0] = *(const v4u*)(base + row * pitch + kb);
  f.q[1] = *(const v4u*)(base + row * pitch + kb + 8);
  return f.v;
}

// ---- B fragment: 32x16 bf16 from packed weights Wt[n][k] (k contiguous) ----
// lane<16 : N=lane, K=k0..k0+15 ; lane>=16 : N=lane-16, K=k0+16..k0+31
__device__ __forceinline__ bf16x16 load_b(const bf16_t* __restrict__ base, int K,
                                          int k0, int lane) {
  int n  = lane & 15;
  int kb = k0 + ((lane >> 4) << 4);
  const v4u* p = (const v4u*)(base + n * K + kb);
  FragU f;
  f.q[0] = p[0];
  f.q[1] = p[1];
  return f.v;
}

__device__ __forceinline__ v8f wmma_bf16(bf16x16 a, bf16x16 b, v8f c) {
  return __builtin_amdgcn_wmma_f32_16x16x32_bf16(false, a, false, b, (short)0, c,
                                                 false, false);
}

// Register-blocked GEMM: 2 m-tiles (rows 0..15, 16..31) x NT n-tiles.
// A fragments shared across all NT n-tiles; B fragments shared across m-tiles.
template <int NT>
__device__ __forceinline__ void gemm_nt(const bf16_t* __restrict__ Ap, int pitch,
                                        const bf16_t* __restrict__ Bp, int bstride,
                                        int K, int lane, v8f acc[][2]) {
#pragma unroll 2
  for (int k0 = 0; k0 < K; k0 += 32) {
    bf16x16 a0 = load_a(Ap, pitch, 0,  k0, lane);
    bf16x16 a1 = load_a(Ap, pitch, 16, k0, lane);
    bf16x16 b[NT];
#pragma unroll
    for (int j = 0; j < NT; ++j) b[j] = load_b(Bp + j * bstride, K, k0, lane);
#pragma unroll
    for (int j = 0; j < NT; ++j) {
      acc[j][0] = wmma_bf16(a0, b[j], acc[j][0]);
      acc[j][1] = wmma_bf16(a1, b[j], acc[j][1]);
    }
  }
}

__device__ __forceinline__ void zero_acc(v8f* a, int cnt) {
  v8f z = {0.f,0.f,0.f,0.f,0.f,0.f,0.f,0.f};
#pragma unroll
  for (int i = 0; i < 8; ++i)
    if (i < cnt) a[i] = z;
}

// ---------------- weight pack: fp32 [K][Nw] -> bf16 [Nw][K] ----------------
__global__ void pack_w(const float* __restrict__ src, bf16_t* __restrict__ dst,
                       int K, int Nw) {
  int idx = blockIdx.x * 256 + threadIdx.x;
  if (idx < K * Nw) {
    int n = idx / K;
    int k = idx - n * K;
    dst[idx] = (bf16_t)src[k * Nw + n];
  }
}

// ---------------- fused gated-MLP kernel ----------------
__global__ __launch_bounds__(256)
void fused_mlp(const float* __restrict__ value,
               const bf16_t* __restrict__ P10,   // W1_0e^T bf16 [512][512]
               const bf16_t* __restrict__ P11,   // W1_1o^T bf16 [256][512]
               const bf16_t* __restrict__ Pm0,   // Wm_0e^T bf16 [512][256]
               const bf16_t* __restrict__ Pm1,   // Wm_1o^T bf16 [256][256]
               const float* __restrict__ bm,     // [512]
               const float* __restrict__ W20,    // [256]
               const float* __restrict__ b2,     // [1]
               const float* __restrict__ W21,    // [256]
               float* __restrict__ out) {
  __shared__ __align__(16) bf16_t sX[4 * X_PLANE];   // X0, X1[0..2]  (later: S0,S1[0..2])
  __shared__ __align__(16) bf16_t sH[4 * H_PLANE];   // H0, H1[0..2]
  __shared__ __align__(16) float  sG[ROWS_PER_WG * H_PITCH]; // gates (fp32); later reduce scratch

  const int tid  = threadIdx.x;
  const int lane = tid & 31;
  const int wv   = tid >> 5;
  const int wg   = blockIdx.x;

  // ---------- stage X: fp32 -> bf16, deinterleave (i,c) ----------
  for (int m = 0; m < ROWS_PER_WG; ++m) {
    const float* src = value + ((size_t)(wg * ROWS_PER_WG + m)) * 2048;
#pragma unroll
    for (int part = 0; part < 2; ++part) {
      int idx = part * 256 + tid;         // 0..511 float4 per row
      float4 v4 = *(const float4*)(src + idx * 4);
      float vv[4] = {v4.x, v4.y, v4.z, v4.w};
      int c2 = idx * 4;
      if (c2 < MUL_IN) {
#pragma unroll
        for (int e = 0; e < 4; ++e)
          sX[m * X_PITCH + c2 + e] = (bf16_t)vv[e];
      } else {
#pragma unroll
        for (int e = 0; e < 4; ++e) {
          int q = c2 - MUL_IN + e;
          int i = q / 3;
          int c = q - 3 * i;
          sX[(1 + c) * X_PLANE + m * X_PITCH + i] = (bf16_t)vv[e];
        }
      }
    }
  }
  __syncthreads();

  // ---------- phase A: Y0 = X0 @ W1_0e ; gate ----------
  {                                                  // n-tiles wv, wv+8, wv+16, wv+24
    v8f acc[4][2];
    zero_acc(&acc[0][0], 8);
    gemm_nt<4>(sX, X_PITCH, P10 + wv * 16 * MUL_IN, 8 * 16 * MUL_IN, MUL_IN,
               lane, acc);
    int mbase = (lane >> 4) << 3;
#pragma unroll
    for (int j = 0; j < 4; ++j) {
      int n = (wv + 8 * j) * 16 + (lane & 15);
      if (j < 2) {                                   // t<16 : silu -> H0 (bf16)
#pragma unroll
        for (int r = 0; r < 8; ++r) {
          sH[(mbase + r) * H_PITCH + n]      = (bf16_t)(silu_f(acc[j][0][r] * A_IN) * C_SILU);
          sH[(16 + mbase + r) * H_PITCH + n] = (bf16_t)(silu_f(acc[j][1][r] * A_IN) * C_SILU);
        }
      } else {                                       // sigmoid gates -> sG (fp32)
        int g = n - MUL_H;
#pragma unroll
        for (int r = 0; r < 8; ++r) {
          sG[(mbase + r) * H_PITCH + g]      = sigmoid_f(acc[j][0][r] * A_IN) * C_SIG;
          sG[(16 + mbase + r) * H_PITCH + g] = sigmoid_f(acc[j][1][r] * A_IN) * C_SIG;
        }
      }
    }
  }
  __syncthreads();

  // ---------- phase B: Y1c = X1c @ W1_1o ; H1 = g * Y1 ----------
  for (int p = 0; p < 3; ++p) {                      // n-tiles wv, wv+8 per plane
    v8f acc[2][2];
    zero_acc(&acc[0][0], 4);
    gemm_nt<2>(sX + (1 + p) * X_PLANE, X_PITCH, P11 + wv * 16 * MUL_IN,
               8 * 16 * MUL_IN, MUL_IN, lane, acc);
    int mbase = (lane >> 4) << 3;
#pragma unroll
    for (int j = 0; j < 2; ++j) {
      int n = (wv + 8 * j) * 16 + (lane & 15);
#pragma unroll
      for (int r = 0; r < 8; ++r) {
        int m0 = mbase + r, m1 = 16 + mbase + r;
        sH[(1 + p) * H_PLANE + m0 * H_PITCH + n] =
            (bf16_t)(sG[m0 * H_PITCH + n] * (acc[j][0][r] * A_IN));
        sH[(1 + p) * H_PLANE + m1 * H_PITCH + n] =
            (bf16_t)(sG[m1 * H_PITCH + n] * (acc[j][1][r] * A_IN));
      }
    }
  }
  __syncthreads();

  // ---------- phase C: Z0 = H0 @ Wm_0e + bm ; gate ----------
  bf16_t* sS = sX;                                   // S planes reuse X region
  {
    v8f acc[4][2];
    zero_acc(&acc[0][0], 8);
    gemm_nt<4>(sH, H_PITCH, Pm0 + wv * 16 * MUL_H, 8 * 16 * MUL_H, MUL_H,
               lane, acc);
    int mbase = (lane >> 4) << 3;
#pragma unroll
    for (int j = 0; j < 4; ++j) {
      int n = (wv + 8 * j) * 16 + (lane & 15);
      float bmv = bm[n];
      if (j < 2) {
#pragma unroll
        for (int r = 0; r < 8; ++r) {
          sS[(mbase + r) * H_PITCH + n] =
              (bf16_t)(silu_f(acc[j][0][r] * A_H + bmv) * C_SILU);
          sS[(16 + mbase + r) * H_PITCH + n] =
              (bf16_t)(silu_f(acc[j][1][r] * A_H + bmv) * C_SILU);
        }
      } else {
        int g = n - MUL_H;
#pragma unroll
        for (int r = 0; r < 8; ++r) {
          sG[(mbase + r) * H_PITCH + g]      = sigmoid_f(acc[j][0][r] * A_H + bmv) * C_SIG;
          sG[(16 + mbase + r) * H_PITCH + g] = sigmoid_f(acc[j][1][r] * A_H + bmv) * C_SIG;
        }
      }
    }
  }
  __syncthreads();

  // ---------- phase D: Z1c = H1c @ Wm_1o ; S1 = g2 * Z1 ----------
  for (int p = 0; p < 3; ++p) {
    v8f acc[2][2];
    zero_acc(&acc[0][0], 4);
    gemm_nt<2>(sH + (1 + p) * H_PLANE, H_PITCH, Pm1 + wv * 16 * MUL_H,
               8 * 16 * MUL_H, MUL_H, lane, acc);
    int mbase = (lane >> 4) << 3;
#pragma unroll
    for (int j = 0; j < 2; ++j) {
      int n = (wv + 8 * j) * 16 + (lane & 15);
#pragma unroll
      for (int r = 0; r < 8; ++r) {
        int m0 = mbase + r, m1 = 16 + mbase + r;
        sS[(1 + p) * H_PLANE + m0 * H_PITCH + n] =
            (bf16_t)(sG[m0 * H_PITCH + n] * (acc[j][0][r] * A_H));
        sS[(1 + p) * H_PLANE + m1 * H_PITCH + n] =
            (bf16_t)(sG[m1 * H_PITCH + n] * (acc[j][1][r] * A_H));
      }
    }
  }
  __syncthreads();

  // ---------- phase E: readout dots (fp32 VALU + LDS reduce) ----------
  {
    int r   = tid >> 3;                 // row 0..31
    int seg = tid & 7;                  // 32-wide K segment
    float p0 = 0.f, p1 = 0.f, p2 = 0.f, p3 = 0.f;
#pragma unroll 8
    for (int j = 0; j < 32; ++j) {
      int k = seg * 32 + j;
      float w0 = W20[k];
      float w1 = W21[k];
      p0 += (float)sS[r * H_PITCH + k] * w0;
      p1 += (float)sS[1 * H_PLANE + r * H_PITCH + k] * w1;
      p2 += (float)sS[2 * H_PLANE + r * H_PITCH + k] * w1;
      p3 += (float)sS[3 * H_PLANE + r * H_PITCH + k] * w1;
    }
    float* red = sG;                    // reuse gate buffer as reduce scratch
    red[(tid)*4 + 0] = p0;
    red[(tid)*4 + 1] = p1;
    red[(tid)*4 + 2] = p2;
    red[(tid)*4 + 3] = p3;
  }
  __syncthreads();
  if (tid < ROWS_PER_WG) {
    const float* red = sG;
    float o0 = 0.f, o1 = 0.f, o2 = 0.f, o3 = 0.f;
#pragma unroll
    for (int s = 0; s < 8; ++s) {
      int b = (tid * 8 + s) * 4;
      o0 += red[b + 0];
      o1 += red[b + 1];
      o2 += red[b + 2];
      o3 += red[b + 3];
    }
    size_t ob = ((size_t)wg * ROWS_PER_WG + tid) * 4;
    out[ob + 0] = o0 * A_H + b2[0];
    out[ob + 1] = o1 * A_H;
    out[ob + 2] = o2 * A_H;
    out[ob + 3] = o3 * A_H;
  }
}

// ---------------- host launch ----------------
extern "C" void kernel_launch(void* const* d_in, const int* in_sizes, int n_in,
                              void* d_out, int out_size, void* d_ws, size_t ws_size,
                              hipStream_t stream) {
  (void)in_sizes; (void)n_in; (void)out_size; (void)ws_size;
  const float* value = (const float*)d_in[0];
  const float* W1_0e = (const float*)d_in[1];
  const float* W1_1o = (const float*)d_in[2];
  const float* Wm_0e = (const float*)d_in[3];
  const float* bm_0e = (const float*)d_in[4];
  const float* Wm_1o = (const float*)d_in[5];
  const float* W2_0e = (const float*)d_in[6];
  const float* b2_0e = (const float*)d_in[7];
  const float* W2_1o = (const float*)d_in[8];

  bf16_t* P10 = (bf16_t*)d_ws;                       // [512][512]
  bf16_t* P11 = P10 + 512 * 512;                     // [256][512]
  bf16_t* Pm0 = P11 + 256 * 512;                     // [512][256]
  bf16_t* Pm1 = Pm0 + 512 * 256;                     // [256][256]

  pack_w<<<(512 * 512 + 255) / 256, 256, 0, stream>>>(W1_0e, P10, 512, 512);
  pack_w<<<(512 * 256 + 255) / 256, 256, 0, stream>>>(W1_1o, P11, 512, 256);
  pack_w<<<(256 * 512 + 255) / 256, 256, 0, stream>>>(Wm_0e, Pm0, 256, 512);
  pack_w<<<(256 * 256 + 255) / 256, 256, 0, stream>>>(Wm_1o, Pm1, 256, 256);

  fused_mlp<<<N_ROWS / ROWS_PER_WG, 256, 0, stream>>>(
      value, P10, P11, Pm0, Pm1, bm_0e, W2_0e, b2_0e, W2_1o, (float*)d_out);
}
